// TemporalFocalTransformerBlock_21260088115594
// MI455X (gfx1250) — matrix operand order repaired and checked by reading, not compiled, not run on hardware
//
#include <hip/hip_runtime.h>
#include <cstddef>
#include <cstdint>

typedef __attribute__((ext_vector_type(16))) __bf16 v16bf;
typedef __attribute__((ext_vector_type(8)))  float  v8f;

union AF { v16bf v; uint4 q[2]; unsigned short u[16]; };

__device__ __forceinline__ unsigned short f2bf(float f) {
  unsigned int u = __float_as_uint(f);
  u = (u + 0x7FFFu + ((u >> 16) & 1u)) >> 16;
  return (unsigned short)u;
}
__device__ __forceinline__ float bf2f(unsigned short h) {
  return __uint_as_float(((unsigned int)h) << 16);
}

// ---------------- LayerNorm 1: x -> xn (bf16) ----------------
__global__ __launch_bounds__(256) void k_ln1(const float* __restrict__ x,
                                             const float* __restrict__ g,
                                             const float* __restrict__ be,
                                             unsigned short* __restrict__ out) {
  __shared__ float red[256];
  int row = blockIdx.x, tid = threadIdx.x;
  const float* xr = x + (size_t)row * 512;
  float v0 = xr[tid], v1 = xr[tid + 256];
  red[tid] = v0 + v1; __syncthreads();
  for (int s = 128; s > 0; s >>= 1) { if (tid < s) red[tid] += red[tid + s]; __syncthreads(); }
  float mu = red[0] * (1.f / 512.f); __syncthreads();
  float d0 = v0 - mu, d1 = v1 - mu;
  red[tid] = d0 * d0 + d1 * d1; __syncthreads();
  for (int s = 128; s > 0; s >>= 1) { if (tid < s) red[tid] += red[tid + s]; __syncthreads(); }
  float rstd = rsqrtf(red[0] * (1.f / 512.f) + 1e-5f);
  unsigned short* o = out + (size_t)row * 512;
  o[tid]       = f2bf(d0 * rstd * g[tid] + be[tid]);
  o[tid + 256] = f2bf(d1 * rstd * g[tid + 256] + be[tid + 256]);
}

// ---------------- weight pack: (K x N) f32 -> (Npad x Kpad) bf16 transposed, zero pad ----------------
__global__ __launch_bounds__(256) void k_packw(const float* __restrict__ in,
                                               unsigned short* __restrict__ out,
                                               int N, int K, int Kpad, long total) {
  long idx = (long)blockIdx.x * 256 + threadIdx.x;
  if (idx >= total) return;
  int kk = (int)(idx % Kpad);
  int n  = (int)(idx / Kpad);
  float v = (n < N && kk < K) ? in[(size_t)kk * N + n] : 0.f;
  out[(size_t)n * Kpad + kk] = f2bf(v);
}

// ---------------- pooled tokens: window-mean of xn, rows (b,t,hw,ww) ----------------
__global__ __launch_bounds__(256) void k_pooled(const unsigned short* __restrict__ xn,
                                                const float* __restrict__ wpool,
                                                const float* __restrict__ bpool,
                                                unsigned short* __restrict__ pooled) {
  int idx = blockIdx.x * 256 + threadIdx.x;   // 256*512
  int c = idx & 511; int row = idx >> 9;
  int ww = row & 3, hw = (row >> 2) & 3; int bt = row >> 4;
  float s = bpool[0];
  for (int wi = 0; wi < 5; ++wi)
    for (int wj = 0; wj < 9; ++wj) {
      int h = hw * 5 + wi, xx = ww * 9 + wj;
      s += bf2f(xn[(((size_t)bt * 20 + h) * 36 + xx) * 512 + c]) * wpool[wi * 9 + wj];
    }
  pooled[(size_t)row * 512 + c] = f2bf(s);
}

// ---------------- bf16 WMMA GEMM, 32x64 register block per wave ----------------
// C = A(M x K) * Bt(N x K)^T + bias [+resid]; Mt2 = Mtiles/2, Nt4 = Ntiles/4.
__global__ __launch_bounds__(256) void k_gemm(const unsigned short* __restrict__ A, int lda,
                                              const unsigned short* __restrict__ Bt, int ldb,
                                              const float* __restrict__ bias, int nBias,
                                              const float* __restrict__ resid,
                                              float* __restrict__ outF,
                                              unsigned short* __restrict__ outB, int ldc,
                                              int Mt2, int Nt4, int K) {
  int wid = threadIdx.x >> 5, lane = threadIdx.x & 31;
  int tile = blockIdx.x * 8 + wid;
  if (tile >= Mt2 * Nt4) return;
  int tm = (tile / Nt4) * 2;
  int tn = (tile % Nt4) * 4;
  int l16 = lane & 15; int hi = (lane >> 4) & 1;
  v8f acc[2][4];
#pragma unroll
  for (int mi = 0; mi < 2; ++mi)
#pragma unroll
    for (int j = 0; j < 4; ++j)
#pragma unroll
      for (int v = 0; v < 8; ++v) acc[mi][j][v] = 0.f;
  const unsigned short* aRow0 = A + (size_t)(tm * 16 + l16) * lda + hi * 8;
  const unsigned short* aRow1 = aRow0 + (size_t)16 * lda;
  const unsigned short* bRow[4];
#pragma unroll
  for (int j = 0; j < 4; ++j)
    bRow[j] = Bt + (size_t)((tn + j) * 16 + l16) * ldb + hi * 8;
  for (int k0 = 0; k0 < K; k0 += 32) {
    AF a0, a1, b[4];
    a0.q[0] = *(const uint4*)(aRow0 + k0);
    a0.q[1] = *(const uint4*)(aRow0 + k0 + 16);
    a1.q[0] = *(const uint4*)(aRow1 + k0);
    a1.q[1] = *(const uint4*)(aRow1 + k0 + 16);
#pragma unroll
    for (int j = 0; j < 4; ++j) {
      b[j].q[0] = *(const uint4*)(bRow[j] + k0);
      b[j].q[1] = *(const uint4*)(bRow[j] + k0 + 16);
    }
#pragma unroll
    for (int j = 0; j < 4; ++j) {
      acc[0][j] = __builtin_amdgcn_wmma_f32_16x16x32_bf16(false, a0.v, false, b[j].v, (short)0, acc[0][j], false, false);
      acc[1][j] = __builtin_amdgcn_wmma_f32_16x16x32_bf16(false, a1.v, false, b[j].v, (short)0, acc[1][j], false, false);
    }
  }
#pragma unroll
  for (int j = 0; j < 4; ++j) {
    int col = (tn + j) * 16 + l16;
    float bv = (col < nBias) ? bias[col] : 0.f;
#pragma unroll
    for (int mi = 0; mi < 2; ++mi) {
#pragma unroll
      for (int v = 0; v < 8; ++v) {
        int rowi = (tm + mi) * 16 + v + hi * 8;
        size_t o = (size_t)rowi * ldc + col;
        float val = acc[mi][j][v] + bv;
        if (resid) val += resid[o];
        if (outF) outF[o] = val;
        else outB[o] = f2bf(val);
      }
    }
  }
}

// ---------------- build scaled windowed Q (pad rows 360..367 = 0) ----------------
__global__ __launch_bounds__(256) void k_buildq(const unsigned short* __restrict__ qkv,
                                                unsigned short* __restrict__ qbuf) {
  long idx = (long)blockIdx.x * 256 + threadIdx.x;  // 32*4*368*128
  int d = idx & 127; long r = idx >> 7;
  int j = (int)(r % 368); r /= 368;
  int n = (int)(r & 3); int w = (int)(r >> 2);
  unsigned short outv = 0;
  if (j < 360) {
    int t = j / 45, pos = j % 45;
    int wi = pos / 9, wj = pos % 9;
    int b = w >> 4, hw = (w >> 2) & 3, ww = w & 3;
    int h = hw * 5 + wi, xx = ww * 9 + wj;
    size_t tok = ((size_t)(b * 8 + t) * 20 + h) * 36 + xx;
    outv = f2bf(bf2f(qkv[tok * 1536 + n * 128 + d]) * 0.08838834764831845f);
  }
  qbuf[idx] = outv;
}

// ---------------- build K (row-major [kv][d]) and V^T ([d][kv]) incl. rolled + pooled parts ----------------
__global__ __launch_bounds__(256) void k_buildkv(const unsigned short* __restrict__ qkv,
                                                 const unsigned short* __restrict__ qkvp,
                                                 unsigned short* __restrict__ kbuf,
                                                 unsigned short* __restrict__ vbufT) {
  long idx = (long)blockIdx.x * 256 + threadIdx.x;  // 32*4*1680*128
  int d = idx & 127; long r = idx >> 7;
  int j = (int)(r % 1680); r /= 1680;
  int n = (int)(r & 3); int w = (int)(r >> 2);
  int b = w >> 4, hw = (w >> 2) & 3, ww = w & 3;
  float kv = 0.f, vv = 0.f;
  if (j < 1320) {
    int t, h, xx;
    if (j < 360) {
      t = j / 45; int pos = j % 45;
      h = hw * 5 + pos / 9; xx = ww * 9 + pos % 9;
    } else {
      int p = j - 360; t = p / 120; int sIdx = p % 120;
      int part = sIdx / 30, s2 = sIdx % 30;
      int rr, cc;
      if (part == 0)      { if (s2 < 12) { rr = s2 / 4; cc = 5 + (s2 & 3); } else { int u = s2 - 12; rr = 3 + u / 9; cc = u % 9; } }
      else if (part == 1) { if (s2 < 12) { rr = s2 / 4; cc = s2 & 3; }      else { int u = s2 - 12; rr = 3 + u / 9; cc = u % 9; } }
      else if (part == 2) { if (s2 < 18) { rr = s2 / 9; cc = s2 % 9; }      else { int u = s2 - 18; rr = 2 + u / 4; cc = 5 + (u & 3); } }
      else                { if (s2 < 18) { rr = s2 / 9; cc = s2 % 9; }      else { int u = s2 - 18; rr = 2 + u / 4; cc = u & 3; } }
      int sh = (part < 2) ? -2 : 2;
      int sw = ((part & 1) == 0) ? -4 : 4;
      h  = ((hw * 5 + rr - sh) % 20 + 20) % 20;
      xx = ((ww * 9 + cc - sw) % 36 + 36) % 36;
    }
    size_t tok = ((size_t)(b * 8 + t) * 20 + h) * 36 + xx;
    kv = bf2f(qkv[tok * 1536 + 512 + n * 128 + d]);
    vv = bf2f(qkv[tok * 1536 + 1024 + n * 128 + d]);
  } else {
    int p = j - 1320; int t = p / 45; int kk = p % 45;
    int ki = kk / 9, kj = kk % 9;
    int y = hw + ki - 2, xp = ww + kj - 4;
    if (y >= 0 && y < 4 && xp >= 0 && xp < 4) {
      size_t prow = (size_t)(b * 8 + t) * 16 + y * 4 + xp;
      kv = bf2f(qkvp[prow * 1536 + 512 + n * 128 + d]);
      vv = bf2f(qkvp[prow * 1536 + 1024 + n * 128 + d]);
    }
  }
  kbuf[((size_t)(w * 4 + n) * 1680 + j) * 128 + d] = f2bf(kv);
  vbufT[((size_t)(w * 4 + n) * 128 + d) * 1696 + j] = f2bf(vv);
}

__global__ __launch_bounds__(256) void k_zerov(unsigned short* __restrict__ vbufT) {
  int idx = blockIdx.x * 256 + threadIdx.x;   // 32*4*128*16
  int j = 1680 + (idx & 15);
  int r = idx >> 4;
  vbufT[(size_t)r * 1696 + j] = 0;
}

// ---------------- flash attention: one wave per (window, head, 16-row q tile) ----------------
__global__ __launch_bounds__(256) void k_attn(const unsigned short* __restrict__ qbuf,
                                              const unsigned short* __restrict__ kbuf,
                                              const unsigned short* __restrict__ vbufT,
                                              unsigned short* __restrict__ attn_out) {
  __shared__ __align__(16) unsigned short plds[8][512];
  int wid = threadIdx.x >> 5, lane = threadIdx.x & 31;
  int gw = blockIdx.x * 8 + wid;               // 0..2943 exact
  int qt = gw % 23; int wn = gw / 23;
  int n = wn & 3; int w = wn >> 2;
  int l16 = lane & 15; int hi = (lane >> 4) & 1;
  const unsigned short* qbase = qbuf + ((size_t)(w * 4 + n) * 368 + qt * 16 + l16) * 128 + hi * 8;
  AF qa[4];
#pragma unroll
  for (int kc = 0; kc < 4; ++kc) {
    qa[kc].q[0] = *(const uint4*)(qbase + kc * 32);
    qa[kc].q[1] = *(const uint4*)(qbase + kc * 32 + 16);
  }
  v8f O[8];
  float rm_[8], rl_[8];
#pragma unroll
  for (int j = 0; j < 8; ++j)
#pragma unroll
    for (int v = 0; v < 8; ++v) O[j][v] = 0.f;
#pragma unroll
  for (int v = 0; v < 8; ++v) { rm_[v] = -1e30f; rl_[v] = 0.f; }
  const unsigned short* kbase = kbuf + (size_t)(w * 4 + n) * 1680 * 128 + hi * 8;
  const unsigned short* vbase = vbufT + (size_t)(w * 4 + n) * 128 * 1696;
  int oi = (w >> 2) & 3, oj = w & 3;
  unsigned short* pl = plds[wid];
  for (int kt = 0; kt < 105; ++kt) {
    v8f s;
#pragma unroll
    for (int v = 0; v < 8; ++v) s[v] = 0.f;
    const unsigned short* kr = kbase + (size_t)(kt * 16 + l16) * 128;
#pragma unroll
    for (int kc = 0; kc < 4; ++kc) {
      AF kb;
      kb.q[0] = *(const uint4*)(kr + kc * 32);
      kb.q[1] = *(const uint4*)(kr + kc * 32 + 16);
      s = __builtin_amdgcn_wmma_f32_16x16x32_bf16(false, qa[kc].v, false, kb.v, (short)0, s, false, false);
    }
    int col = kt * 16 + l16;
    float mk = 0.f;
    if (col >= 1320) {
      int kk = (col - 1320) % 45;
      int ki = kk / 9, kj = kk % 9;
      int y = oi + ki - 2, xp = oj + kj - 4;
      if (!(y >= 0 && y < 4 && xp >= 0 && xp < 4)) mk = -100.f;
    }
    float pvv[8];
#pragma unroll
    for (int v = 0; v < 8; ++v) {
      float sv = s[v] + mk;
      float rmax = sv;
      rmax = fmaxf(rmax, __shfl_xor(rmax, 1));
      rmax = fmaxf(rmax, __shfl_xor(rmax, 2));
      rmax = fmaxf(rmax, __shfl_xor(rmax, 4));
      rmax = fmaxf(rmax, __shfl_xor(rmax, 8));
      float mn = fmaxf(rm_[v], rmax);
      float sc = __expf(rm_[v] - mn);
      float p  = __expf(sv - mn);
      float rs = p;
      rs += __shfl_xor(rs, 1); rs += __shfl_xor(rs, 2);
      rs += __shfl_xor(rs, 4); rs += __shfl_xor(rs, 8);
      rl_[v] = rl_[v] * sc + rs;
      rm_[v] = mn;
#pragma unroll
      for (int j = 0; j < 8; ++j) O[j][v] *= sc;
      pvv[v] = p;
    }
    int cbase = (kt & 1) * 16;
#pragma unroll
    for (int v = 0; v < 8; ++v)
      pl[(v + hi * 8) * 32 + cbase + l16] = f2bf(pvv[v]);
    bool tail = (kt == 104);
    if (tail && lane < 16) {
      *(uint4*)&pl[lane * 32 + 16] = make_uint4(0, 0, 0, 0);
      *(uint4*)&pl[lane * 32 + 24] = make_uint4(0, 0, 0, 0);
    }
    if ((kt & 1) || tail) {
      asm volatile("s_wait_dscnt 0" ::: "memory");
      AF pa;
      pa.q[0] = *(const uint4*)&pl[l16 * 32 + hi * 8];
      pa.q[1] = *(const uint4*)&pl[l16 * 32 + 16 + hi * 8];
      int kv0 = (kt >> 1) * 32;
#pragma unroll
      for (int j = 0; j < 8; ++j) {
        const unsigned short* vr = vbase + (size_t)(j * 16 + l16) * 1696 + kv0 + hi * 8;
        AF vb;
        vb.q[0] = *(const uint4*)(vr);
        vb.q[1] = *(const uint4*)(vr + 16);
        O[j] = __builtin_amdgcn_wmma_f32_16x16x32_bf16(false, pa.v, false, vb.v, (short)0, O[j], false, false);
      }
    }
  }
#pragma unroll
  for (int v = 0; v < 8; ++v) {
    int qrow = qt * 16 + v + hi * 8;
    if (qrow >= 360) continue;
    float inv = 1.f / rl_[v];
    size_t ro = (size_t)(w * 360 + qrow) * 512 + n * 128;
#pragma unroll
    for (int j = 0; j < 8; ++j)
      attn_out[ro + j * 16 + l16] = f2bf(O[j][v] * inv);
  }
}

// ---------------- residual + LayerNorm 2 (un-windows proj rows) ----------------
__global__ __launch_bounds__(256) void k_resid_ln(const float* __restrict__ x,
                                                  const float* __restrict__ proj,
                                                  const float* __restrict__ g,
                                                  const float* __restrict__ be,
                                                  float* __restrict__ x1,
                                                  unsigned short* __restrict__ y) {
  __shared__ float red[256];
  int row = blockIdx.x, tid = threadIdx.x;
  int tmp = row;
  int wd = tmp % 36; tmp /= 36;
  int h  = tmp % 20; tmp /= 20;
  int t  = tmp % 8;  int b = tmp / 8;
  int hw = h / 5, wi = h % 5, ww = wd / 9, wj = wd % 9;
  int w = (b * 4 + hw) * 4 + ww;
  size_t prow = (size_t)(w * 360 + t * 45 + wi * 9 + wj) * 512;
  const float* xr = x + (size_t)row * 512;
  float v0 = xr[tid] + proj[prow + tid];
  float v1 = xr[tid + 256] + proj[prow + tid + 256];
  float* xo = x1 + (size_t)row * 512;
  xo[tid] = v0; xo[tid + 256] = v1;
  red[tid] = v0 + v1; __syncthreads();
  for (int s = 128; s > 0; s >>= 1) { if (tid < s) red[tid] += red[tid + s]; __syncthreads(); }
  float mu = red[0] * (1.f / 512.f); __syncthreads();
  float d0 = v0 - mu, d1 = v1 - mu;
  red[tid] = d0 * d0 + d1 * d1; __syncthreads();
  for (int s = 128; s > 0; s >>= 1) { if (tid < s) red[tid] += red[tid + s]; __syncthreads(); }
  float rstd = rsqrtf(red[0] * (1.f / 512.f) + 1e-5f);
  unsigned short* yo = y + (size_t)row * 512;
  yo[tid]       = f2bf(d0 * rstd * g[tid] + be[tid]);
  yo[tid + 256] = f2bf(d1 * rstd * g[tid + 256] + be[tid + 256]);
}

// ---------------- T2T fold: gather <=9 patch contributions per output pixel (h ld = 1984) ----------------
__global__ __launch_bounds__(256) void k_fold(const unsigned short* __restrict__ h,
                                              float* __restrict__ hf) {
  long idx = (long)blockIdx.x * 256 + threadIdx.x;  // 16*40*60*108
  int xx = (int)(idx % 108); long r = idx / 108;
  int yy = (int)(r % 60); r /= 60;
  int c = (int)(r % 40); int bt = (int)(r / 40);
  float sum = 0.f;
  for (int ki = yy % 3; ki < 7; ki += 3) {
    int ny = yy + 3 - ki;
    if (ny < 0 || ny >= 60) continue;
    int oi = ny / 3;
    for (int kj = xx % 3; kj < 7; kj += 3) {
      int nx = xx + 3 - kj;
      if (nx < 0 || nx >= 108) continue;
      int oj = nx / 3;
      sum += bf2f(h[((size_t)bt * 720 + oi * 36 + oj) * 1984 + c * 49 + ki * 7 + kj]);
    }
  }
  hf[idx] = sum;
}

// ---------------- T2T unfold + normalize + exact GELU -> bf16 (Kpad 1984) ----------------
__global__ __launch_bounds__(256) void k_unfold(const float* __restrict__ hf,
                                                unsigned short* __restrict__ h2) {
  long idx = (long)blockIdx.x * 256 + threadIdx.x;  // 16*720*1984
  int cc = (int)(idx % 1984); long r = idx / 1984;
  int l = (int)(r % 720); int bt = (int)(r / 720);
  float outv = 0.f;
  if (cc < 1960) {
    int c = cc / 49, kk = cc % 49, ki = kk / 7, kj = kk % 7;
    int oi = l / 36, oj = l % 36;
    int y = oi * 3 + ki - 3, xp = oj * 3 + kj - 3;
    if (y >= 0 && y < 60 && xp >= 0 && xp < 108) {
      int cy = 0, cx = 0;
      for (int a = y % 3; a < 7; a += 3) { int nn = y + 3 - a; if (nn >= 0 && nn < 60) ++cy; }
      for (int a = xp % 3; a < 7; a += 3) { int nn = xp + 3 - a; if (nn >= 0 && nn < 108) ++cx; }
      float v = hf[((size_t)(bt * 40 + c) * 60 + y) * 108 + xp] / (float)(cy * cx);
      outv = 0.5f * v * (1.f + erff(v * 0.70710678118654752f));
    }
  }
  h2[idx] = f2bf(outv);
}

extern "C" void kernel_launch(void* const* d_in, const int* in_sizes, int n_in,
                              void* d_out, int out_size, void* d_ws, size_t ws_size,
                              hipStream_t stream) {
  (void)in_sizes; (void)n_in; (void)out_size; (void)ws_size;
  const float* x     = (const float*)d_in[0];
  const float* g1    = (const float*)d_in[1];
  const float* be1   = (const float*)d_in[2];
  const float* wqkv  = (const float*)d_in[3];
  const float* bqkv  = (const float*)d_in[4];
  const float* wproj = (const float*)d_in[5];
  const float* bproj = (const float*)d_in[6];
  const float* wpool = (const float*)d_in[7];
  const float* bpool = (const float*)d_in[8];
  const float* g2    = (const float*)d_in[9];
  const float* be2   = (const float*)d_in[10];
  const float* w1    = (const float*)d_in[11];
  const float* bf1   = (const float*)d_in[12];
  const float* w2    = (const float*)d_in[13];
  const float* bf2   = (const float*)d_in[14];
  float* out = (float*)d_out;

  char* ws = (char*)d_ws;
  size_t off = 0;
  auto alloc = [&](size_t bytes) { size_t o = off; off += (bytes + 255) & ~(size_t)255; return o; };
  // persistent across both phases
  size_t o_x1  = alloc(11520ULL * 512 * 4);
  size_t o_y   = alloc(11520ULL * 512 * 2);
  size_t o_w1T = alloc(1984ULL * 512 * 2);
  size_t o_w2T = alloc(512ULL * 1984 * 2);
  size_t S = off;
  // phase 1 (attention)
  off = S;
  size_t o_xn    = alloc(11520ULL * 512 * 2);
  size_t o_wqkvT = alloc(1536ULL * 512 * 2);
  size_t o_wprojT= alloc(512ULL * 512 * 2);
  size_t o_pool  = alloc(256ULL * 512 * 2);
  size_t o_qkv   = alloc(11520ULL * 1536 * 2);
  size_t o_qkvp  = alloc(256ULL * 1536 * 2);
  size_t o_qbuf  = alloc(32ULL * 4 * 368 * 128 * 2);
  size_t o_kbuf  = alloc(32ULL * 4 * 1680 * 128 * 2);
  size_t o_vbuf  = alloc(32ULL * 4 * 128 * 1696 * 2);
  size_t o_attn  = alloc(11520ULL * 512 * 2);
  size_t o_proj  = alloc(11520ULL * 512 * 4);
  // phase 2 (FFN) aliases the phase-1 region
  off = S;
  size_t o_h  = alloc(11520ULL * 1984 * 2);
  size_t o_hf = alloc(16ULL * 40 * 60 * 108 * 4);
  size_t o_h2 = alloc(11520ULL * 1984 * 2);

  float*          x1f    = (float*)(ws + o_x1);
  unsigned short* yb     = (unsigned short*)(ws + o_y);
  unsigned short* w1T    = (unsigned short*)(ws + o_w1T);
  unsigned short* w2T    = (unsigned short*)(ws + o_w2T);
  unsigned short* xn     = (unsigned short*)(ws + o_xn);
  unsigned short* wqkvT  = (unsigned short*)(ws + o_wqkvT);
  unsigned short* wprojT = (unsigned short*)(ws + o_wprojT);
  unsigned short* pooledb= (unsigned short*)(ws + o_pool);
  unsigned short* qkvb   = (unsigned short*)(ws + o_qkv);
  unsigned short* qkvpb  = (unsigned short*)(ws + o_qkvp);
  unsigned short* qbuf   = (unsigned short*)(ws + o_qbuf);
  unsigned short* kbuf   = (unsigned short*)(ws + o_kbuf);
  unsigned short* vbufT  = (unsigned short*)(ws + o_vbuf);
  unsigned short* attnb  = (unsigned short*)(ws + o_attn);
  float*          projf  = (float*)(ws + o_proj);
  unsigned short* hb     = (unsigned short*)(ws + o_h);
  float*          hff    = (float*)(ws + o_hf);
  unsigned short* h2b    = (unsigned short*)(ws + o_h2);

  // ---- phase 1 ----
  k_ln1<<<11520, 256, 0, stream>>>(x, g1, be1, xn);
  k_packw<<<3072, 256, 0, stream>>>(wqkv, wqkvT, 1536, 512, 512, 1536L * 512);
  k_packw<<<1024, 256, 0, stream>>>(wproj, wprojT, 512, 512, 512, 512L * 512);
  k_packw<<<3968, 256, 0, stream>>>(w1, w1T, 1960, 512, 512, 1984L * 512);
  k_packw<<<3968, 256, 0, stream>>>(w2, w2T, 512, 1960, 1984, 512L * 1984);
  k_pooled<<<512, 256, 0, stream>>>(xn, wpool, bpool, pooledb);
  // qkv: M=11520 (360 pairs), N=1536 (24 quads), K=512
  k_gemm<<<1080, 256, 0, stream>>>(xn, 512, wqkvT, 512, bqkv, 1536, nullptr,
                                   nullptr, qkvb, 1536, 360, 24, 512);
  // pooled qkv: M=256 (8 pairs), N=1536 (24 quads)
  k_gemm<<<24, 256, 0, stream>>>(pooledb, 512, wqkvT, 512, bqkv, 1536, nullptr,
                                 nullptr, qkvpb, 1536, 8, 24, 512);
  k_buildq<<<94208, 256, 0, stream>>>(qkvb, qbuf);
  k_buildkv<<<107520, 256, 0, stream>>>(qkvb, qkvpb, kbuf, vbufT);
  k_zerov<<<1024, 256, 0, stream>>>(vbufT);
  k_attn<<<368, 256, 0, stream>>>(qbuf, kbuf, vbufT, attnb);
  // proj: M=11520 (360 pairs), N=512 (8 quads)
  k_gemm<<<360, 256, 0, stream>>>(attnb, 512, wprojT, 512, bproj, 512, nullptr,
                                  projf, nullptr, 512, 360, 8, 512);
  k_resid_ln<<<11520, 256, 0, stream>>>(x, projf, g2, be2, x1f, yb);
  // ---- phase 2 ----
  // ffn1: M=11520 (360 pairs), N=1984 (31 quads), K=512
  k_gemm<<<1395, 256, 0, stream>>>(yb, 512, w1T, 512, bf1, 1960, nullptr,
                                   nullptr, hb, 1984, 360, 31, 512);
  k_fold<<<16200, 256, 0, stream>>>(hb, hff);
  k_unfold<<<89280, 256, 0, stream>>>(hff, h2b);
  // ffn2: M=11520 (360 pairs), N=512 (8 quads), K=1984, fused bias+residual -> out
  k_gemm<<<360, 256, 0, stream>>>(h2b, 1984, w2T, 1984, bf2, 512, x1f,
                                  out, nullptr, 512, 360, 8, 1984);
}
